// SNN_Model_62861141344368
// MI455X (gfx1250) — matrix operand order repaired and checked by reading, not compiled, not run on hardware
//
#include <hip/hip_runtime.h>

// ---------------------------------------------------------------------------
// SNN (3-layer LIF) for MI455X / gfx1250, wave32 + WMMA + TDM.
//
//   cur1[t,b,:] = x[t,b,:] @ W1^T + b1 has no recurrent dependency ->
//   hoisted into one [51200 x 2048] x [2048 x 512] f16-WMMA GEMM (107 GFLOP;
//   x traffic 419MB ~= 18us @ 23.3 TB/s -> roughly balanced, f16 WMMA w/ f32
//   accumulation is the right operating point).  The LIF recurrence couples
//   only across t within a batch row -> one WG owns 16 batch rows for all
//   100 steps, membrane state lives in VGPRs, spikes (exact 0/1 in f16) feed
//   layer-2/3 WMMAs from LDS.  The per-step 32KB cur1 tile is DMA-prefetched
//   one step ahead by the Tensor Data Mover into a double-buffered LDS
//   region (TENSORcnt + workgroup barrier), hiding global latency on the
//   sequential critical path.
// ---------------------------------------------------------------------------

typedef __attribute__((ext_vector_type(16))) _Float16 v16h;
typedef __attribute__((ext_vector_type(8)))  _Float16 v8h;
typedef __attribute__((ext_vector_type(8)))  float    v8f;
typedef __attribute__((ext_vector_type(4)))  unsigned int v4u;
typedef __attribute__((ext_vector_type(8)))  int v8i;
typedef __attribute__((ext_vector_type(4)))  int v4i;

namespace {
constexpr int T_STEPS = 100;
constexpr int BATCH   = 512;
constexpr int NIN     = 2048;
constexpr int H1      = 512;
constexpr int H2      = 256;
constexpr int NOUT    = 10;
constexpr int MTOT    = T_STEPS * BATCH;   // 51200 rows in the big GEMM
constexpr int LDA     = 48;                // LDS A-tile stride (halves); 96B rows keep 16B-aligned b128 chunks
}

#if defined(__gfx1250__) && __has_builtin(__builtin_amdgcn_tensor_load_to_lds) && \
    __has_builtin(__builtin_amdgcn_s_wait_tensorcnt)
#define HAVE_TDM 1
#else
#define HAVE_TDM 0
#endif

__device__ __forceinline__ v8f wmma_f16(v16h a, v16h b, v8f c) {
  // v_wmma_f32_16x16x32_f16  (8 args: neg_a, A, neg_b, B, c_mod, C, reuse_a, reuse_b)
  return __builtin_amdgcn_wmma_f32_16x16x32_f16(false, a, false, b, (short)0, c,
                                                false, false);
}

// A-fragment (16x32 f16): lane holds row M=lane&15; halves 0..15 are
// K = k0..k0+15 with k0 = 0 (lanes 0-15) or 16 (lanes 16-31), matching B.
// Two contiguous 16B chunks -> two ds_load_b128.
__device__ __forceinline__ v16h load_a_frag(const _Float16* rowp, int kk, int lane) {
  const int k0 = kk + ((lane & 16) ? 16 : 0);
  v8h lo = *(const v8h*)(rowp + k0);
  v8h hi = *(const v8h*)(rowp + k0 + 8);
  return __builtin_shufflevector(lo, hi, 0, 1, 2, 3, 4, 5, 6, 7,
                                         8, 9, 10, 11, 12, 13, 14, 15);
}

__device__ __forceinline__ v8h cvt2x4(float4 a, float4 b) {
  v8h r;
  r[0] = (_Float16)a.x; r[1] = (_Float16)a.y; r[2] = (_Float16)a.z; r[3] = (_Float16)a.w;
  r[4] = (_Float16)b.x; r[5] = (_Float16)b.y; r[6] = (_Float16)b.z; r[7] = (_Float16)b.w;
  return r;
}

#if HAVE_TDM
// TDM 1-D copy: nelem f32 elements, global -> LDS, via Tensor DMA Descriptor
// (ISA 08_async_tensor §8): group0 = {ctrl, lds_addr, global_addr, type=2},
// group1 packs data_size=4B, tensor_dim0 = tile_dim0 = nelem, dim1 collapsed.
// This toolchain exposes the 6-arg builtin: (g0, g1, g2, g3, g_extra, cpol).
__device__ __forceinline__ void tdm_load_1d(unsigned int lds_off,
                                            const void* gsrc,
                                            unsigned int nelem) {
  unsigned long long ga = (unsigned long long)(uintptr_t)gsrc;
  v4u g0;
  g0[0] = 1u;                                   // count=1, user descriptor
  g0[1] = lds_off;                              // lds_addr (bytes)
  g0[2] = (unsigned int)ga;                     // global_addr[31:0]
  g0[3] = (unsigned int)(ga >> 32) | (2u << 30);// global_addr[56:32] | type=2
  const unsigned int td0 = nelem, td1 = 1u, tile0 = nelem;
  v8i g1;
  g1[0] = (int)(2u << 16);                                  // data_size=4B
  g1[1] = (int)((td0 & 0xFFFFu) << 16);                     // tensor_dim0[15:0]
  g1[2] = (int)(((td0 >> 16) & 0xFFFFu) | ((td1 & 0xFFFFu) << 16));
  g1[3] = (int)(((td1 >> 16) & 0xFFFFu) | (tile0 << 16));   // tile_dim0
  g1[4] = 0;                                                // tile_dim1/2 = 0
  g1[5] = (int)td0;                                         // dim0_stride[31:0]
  g1[6] = 0;                                                // dim0_stride[47:32]
  g1[7] = 0;
  v4i z4 = {};
  v8i z8 = {};
  __builtin_amdgcn_tensor_load_to_lds(g0, g1, z4, z4, z8, 0);
}
#endif

// --------------------------- weight conversion -----------------------------
__global__ void __launch_bounds__(256) cvt_weights_kernel(
    const float* __restrict__ W1, const float* __restrict__ W2,
    const float* __restrict__ W3, _Float16* __restrict__ W1h,
    _Float16* __restrict__ W2h, _Float16* __restrict__ W3h) {
  const int stride = gridDim.x * blockDim.x;
  const int i0 = blockIdx.x * blockDim.x + threadIdx.x;
  for (int i = i0; i < H1 * NIN; i += stride) W1h[i] = (_Float16)W1[i];
  for (int i = i0; i < H2 * H1; i += stride)  W2h[i] = (_Float16)W2[i];
  for (int i = i0; i < 16 * H2; i += stride) {          // W3 padded 10 -> 16 rows
    int r = i >> 8, c = i & (H2 - 1);
    W3h[i] = (r < NOUT) ? (_Float16)W3[r * H2 + c] : (_Float16)0.0f;
  }
}

// ------------------- big GEMM: cur1 = x @ W1^T + b1 ------------------------
// Block = 256 threads = 8 waves laid out 4(M) x 2(N); tile 128x128, K step 32.
// A (x, fp32) is converted to f16 through a double-buffered LDS tile so the
// staging of tile k+1 overlaps the WMMAs on tile k (one barrier per K step).
// B fragments come straight from W1h (2MB, L2-resident).
__global__ void __launch_bounds__(256) gemm1_kernel(
    const float* __restrict__ x, const _Float16* __restrict__ W1h,
    const float* __restrict__ b1, float* __restrict__ cur1) {
  __shared__ _Float16 At[2][128 * LDA];

  const int tid  = threadIdx.x;
  const int lane = tid & 31;
  const int wave = tid >> 5;
  const int wm   = wave & 3;        // 0..3 -> 32-row slice
  const int wn   = wave >> 2;       // 0..1 -> 64-col slice
  const int m0   = blockIdx.y * 128;
  const int n0   = blockIdx.x * 128;

  v8f acc[2][4] = {};

  // A staging: each thread owns 16 consecutive fp32 of one tile row.
  const int arow = tid >> 1;
  const int acol = (tid & 1) * 16;
  const float* gA = x + (size_t)(m0 + arow) * NIN + acol;

  {  // preload K-tile 0 into buffer 0
    const float4* gp = (const float4*)(gA);
    float4 f0 = gp[0], f1 = gp[1], f2 = gp[2], f3 = gp[3];
    _Float16* lA = At[0] + arow * LDA + acol;
    *(v8h*)(lA)     = cvt2x4(f0, f1);
    *(v8h*)(lA + 8) = cvt2x4(f2, f3);
  }
  __syncthreads();

  int p = 0;
  for (int kk = 0; kk < NIN; kk += 32) {
    const bool more = (kk + 32) < NIN;
    float4 nf0 = {}, nf1 = {}, nf2 = {}, nf3 = {};
    if (more) {                       // fetch next K-tile while computing
      const float4* gp = (const float4*)(gA + kk + 32);
      nf0 = gp[0]; nf1 = gp[1]; nf2 = gp[2]; nf3 = gp[3];
      __builtin_prefetch(gA + kk + 96, 0, 1);   // global_prefetch next+2
    }

    const _Float16* Ab = At[p];
    v16h afr[2];
#pragma unroll
    for (int mi = 0; mi < 2; ++mi)
      afr[mi] = load_a_frag(Ab + (wm * 32 + mi * 16 + (lane & 15)) * LDA, 0, lane);

    // B-fragment (32x16): lane holds column N=lane&15; 16 contiguous K halves
    // starting at kb = kk (+16 for lanes 16-31) of W1h row N  (W1h is [n][k]).
    const int kb = kk + ((lane & 16) ? 16 : 0);
#pragma unroll
    for (int ni = 0; ni < 4; ++ni) {
      const _Float16* bp =
          W1h + (size_t)(n0 + wn * 64 + ni * 16 + (lane & 15)) * NIN + kb;
      v16h bfr = *(const v16h*)bp;
      acc[0][ni] = wmma_f16(afr[0], bfr, acc[0][ni]);
      acc[1][ni] = wmma_f16(afr[1], bfr, acc[1][ni]);
    }

    if (more) {
      _Float16* lA = At[p ^ 1] + arow * LDA + acol;
      *(v8h*)(lA)     = cvt2x4(nf0, nf1);
      *(v8h*)(lA + 8) = cvt2x4(nf2, nf3);
    }
    __syncthreads();
    p ^= 1;
  }

  // C layout (ISA 7.12.2): VGPR r -> M = r (lanes 0-15) / r+8 (lanes 16-31).
#pragma unroll
  for (int ni = 0; ni < 4; ++ni) {
    const int colg = n0 + wn * 64 + ni * 16 + (lane & 15);
    const float bias = b1[colg];
    const int rbase = m0 + wm * 32 + ((lane & 16) ? 8 : 0);
#pragma unroll
    for (int mi = 0; mi < 2; ++mi)
#pragma unroll
      for (int r = 0; r < 8; ++r)
        cur1[(size_t)(rbase + mi * 16 + r) * H1 + colg] = acc[mi][ni][r] + bias;
  }
}

// --------------------- recurrent LIF + layer2/3 WMMA -----------------------
// 32 WGs x 256 threads; WG owns batch rows [16*blk, 16*blk+16) for all T.
// Membrane state in VGPRs; spikes in LDS (f16) feed the WMMAs.  cur1 tiles
// (32KB/step, contiguous) are prefetched one step ahead by the TDM.
__global__ void __launch_bounds__(256) snn_recurrent_kernel(
    const float* __restrict__ cur1, const _Float16* __restrict__ W2h,
    const _Float16* __restrict__ W3h, const float* __restrict__ b2,
    const float* __restrict__ b3, float* __restrict__ out) {
  __shared__ _Float16 s1h[16 * H1];   // 16KB: layer-1 spikes (A of GEMM2)
  __shared__ _Float16 s2h[16 * H2];   // 8KB : layer-2 spikes (A of GEMM3)
#if HAVE_TDM
  __shared__ float c1buf[2][16 * H1]; // 2 x 32KB: TDM double buffer for cur1
#endif

  constexpr float kBeta = 0.95f;
  constexpr float kThr  = 1.0f;

  const int tid  = threadIdx.x;
  const int lane = tid & 31;
  const int wave = tid >> 5;
  const int b0   = blockIdx.x * 16;

  float m1r[32];                      // 16x512 state / 256 threads
#pragma unroll
  for (int j = 0; j < 32; ++j) m1r[j] = 0.0f;
  float m2r[2][8] = {};               // matches GEMM2 C-fragment ownership
  float m3r[8]    = {};               // wave 0 only

  const int  col2a  = wave * 32 + (lane & 15);
  const int  col2b  = col2a + 16;
  const float bias2a = b2[col2a];
  const float bias2b = b2[col2b];
  const int  rofs   = (lane & 16) ? 8 : 0;
  const int  col3   = lane & 15;
  const float bias3 = (col3 < NOUT) ? b3[col3] : 0.0f;

#if HAVE_TDM
  if (wave == 0)                       // kick off DMA of tile t=0
    tdm_load_1d((unsigned int)(uintptr_t)&c1buf[0][0],
                cur1 + (size_t)b0 * H1, 16 * H1);
#endif

  for (int t = 0; t < T_STEPS; ++t) {
#if HAVE_TDM
    if (wave == 0) __builtin_amdgcn_s_wait_tensorcnt(0);  // tile t landed
    __syncthreads();                                      // publish to WG
    const int pb = t & 1;
    if (wave == 0 && (t + 1) < T_STEPS)                   // DMA tile t+1
      tdm_load_1d((unsigned int)(uintptr_t)&c1buf[pb ^ 1][0],
                  cur1 + ((size_t)(t + 1) * BATCH + b0) * H1, 16 * H1);
    const float* c1 = c1buf[pb];
#else
    const float* c1 = cur1 + ((size_t)t * BATCH + b0) * H1;
#endif

    // ---- LIF layer 1 (state in regs, spikes -> LDS) ----
#pragma unroll
    for (int j = 0; j < 32; ++j) {
      const int e = tid + j * 256;                 // coalesced / conflict-free
      float m = m1r[j] * kBeta + c1[e];
      float spk = (m > kThr) ? 1.0f : 0.0f;
      m1r[j] = m - spk * kThr;
      s1h[e] = (_Float16)spk;
    }
    __syncthreads();

    // ---- layer 2: [16 x 512] @ W2^T[512 x 256]; wave owns 32 out cols ----
    v8f a0 = {}, a1 = {};
#pragma unroll 4
    for (int kk = 0; kk < H1; kk += 32) {
      v16h af = load_a_frag(s1h + (lane & 15) * H1, kk, lane);
      const int kb = kk + ((lane & 16) ? 16 : 0);
      v16h bf0 = *(const v16h*)(W2h + (size_t)col2a * H1 + kb);
      v16h bf1 = *(const v16h*)(W2h + (size_t)col2b * H1 + kb);
      a0 = wmma_f16(af, bf0, a0);
      a1 = wmma_f16(af, bf1, a1);
    }
    // LIF layer 2 directly on the C fragments (state stays in regs)
#pragma unroll
    for (int r = 0; r < 8; ++r) {
      const int row = r + rofs;
      float m = m2r[0][r] * kBeta + a0[r] + bias2a;
      float spk = (m > kThr) ? 1.0f : 0.0f;
      m2r[0][r] = m - spk * kThr;
      s2h[row * H2 + col2a] = (_Float16)spk;

      m = m2r[1][r] * kBeta + a1[r] + bias2b;
      spk = (m > kThr) ? 1.0f : 0.0f;
      m2r[1][r] = m - spk * kThr;
      s2h[row * H2 + col2b] = (_Float16)spk;
    }
    __syncthreads();

    // ---- layer 3 on wave 0: [16 x 256] @ W3^T (N padded 10 -> 16) ----
    if (wave == 0) {
      v8f a3 = {};
#pragma unroll
      for (int kk = 0; kk < H2; kk += 32) {
        v16h af = load_a_frag(s2h + (lane & 15) * H2, kk, lane);
        const int kb = kk + ((lane & 16) ? 16 : 0);
        v16h bw = *(const v16h*)(W3h + (size_t)col3 * H2 + kb);
        a3 = wmma_f16(af, bw, a3);
      }
      if (col3 < NOUT) {
#pragma unroll
        for (int r = 0; r < 8; ++r) {
          const int row = r + rofs;
          float m = m3r[r] * kBeta + a3[r] + bias3;
          float spk = (m > kThr) ? 1.0f : 0.0f;
          m3r[r] = m - spk * kThr;
          out[((size_t)t * BATCH + b0 + row) * NOUT + col3] = spk;
        }
      }
    }
    __syncthreads();
  }
}

// ------------------------------- launcher ----------------------------------
extern "C" void kernel_launch(void* const* d_in, const int* in_sizes, int n_in,
                              void* d_out, int out_size, void* d_ws, size_t ws_size,
                              hipStream_t stream) {
  (void)in_sizes; (void)n_in; (void)out_size; (void)ws_size;
  const float* x  = (const float*)d_in[0];
  const float* W1 = (const float*)d_in[1];
  const float* b1 = (const float*)d_in[2];
  const float* W2 = (const float*)d_in[3];
  const float* b2 = (const float*)d_in[4];
  const float* W3 = (const float*)d_in[5];
  const float* b3 = (const float*)d_in[6];
  float* out = (float*)d_out;

  // workspace carve-up (all offsets 256B-aligned)
  char* ws = (char*)d_ws;
  _Float16* W1h = (_Float16*)ws;                       // 2 MB
  _Float16* W2h = W1h + (size_t)H1 * NIN;              // 256 KB
  _Float16* W3h = W2h + (size_t)H2 * H1;               // 8 KB (padded 16 rows)
  size_t half_bytes = ((size_t)H1 * NIN + (size_t)H2 * H1 + 16 * H2) * sizeof(_Float16);
  half_bytes = (half_bytes + 255) & ~(size_t)255;
  float* cur1 = (float*)(ws + half_bytes);             // 100 MB: [T*B, 512]

  cvt_weights_kernel<<<dim3(512), dim3(256), 0, stream>>>(W1, W2, W3, W1h, W2h, W3h);

  gemm1_kernel<<<dim3(H1 / 128, MTOT / 128), dim3(256), 0, stream>>>(x, W1h, b1, cur1);

  snn_recurrent_kernel<<<dim3(BATCH / 16), dim3(256), 0, stream>>>(cur1, W2h, W3h,
                                                                   b2, b3, out);
}
